// MHSA_14465449853197
// MI455X (gfx1250) — matrix operand order repaired and checked by reading, not compiled, not run on hardware
//
#include <hip/hip_runtime.h>
#include <hip/hip_bf16.h>

typedef __bf16 bf16_t;
typedef __attribute__((ext_vector_type(16))) __bf16 v16bf;
typedef __attribute__((ext_vector_type(8)))  __bf16 v8bf;
typedef __attribute__((ext_vector_type(8)))  float  v8f;

#define BATCH 4
#define SEQ   2048
#define DIM   1024
#define HEADS 16
#define HD    64
#define MROWS (BATCH * SEQ)   // 8192
#define N3    (3 * DIM)       // 3072

union Frag16 { v16bf v; v8bf h[2]; };

// ---- fragment loaders (ISA 7.12.2 layouts, wave32) -------------------------
// A (16x32 bf16): lane r (<16): K{0..7,16..23}; lane r+16: K{8..15,24..31}
__device__ __forceinline__ v16bf load_a_frag(const bf16_t* base, int ld, int k0, int lane) {
  const int r  = lane & 15;
  const int hi = lane >> 4;
  const bf16_t* p = base + (size_t)r * ld + k0 + hi * 8;
  Frag16 f;
  f.h[0] = *(const v8bf*)(p);
  f.h[1] = *(const v8bf*)(p + 16);
  return f.v;
}

// B (32x16 bf16) from B^T stored row-major [N][K]:
// lane r (<16): col r, K{0..7,16..23} ; lane r+16: col r, K{8..15,24..31}
__device__ __forceinline__ v16bf load_bt_frag(const bf16_t* baseT, int ld, int k0, int lane) {
  const int r  = lane & 15;
  const int hi = lane >> 4;
  const bf16_t* p = baseT + (size_t)r * ld + k0 + hi * 8;
  Frag16 f;
  f.h[0] = *(const v8bf*)(p);
  f.h[1] = *(const v8bf*)(p + 16);
  return f.v;
}

__device__ __forceinline__ v8f wmma_bf16(v16bf a, v16bf b, v8f c) {
  return __builtin_amdgcn_wmma_f32_16x16x32_bf16(false, a, false, b, (short)0, c, false, false);
}

// 8 independent WMMAs of one 32x64 k-step
__device__ __forceinline__ void mma8(v8f c[2][4], v16bf a0, v16bf a1,
                                     v16bf b0, v16bf b1, v16bf b2, v16bf b3) {
  c[0][0] = wmma_bf16(a0, b0, c[0][0]);
  c[1][0] = wmma_bf16(a1, b0, c[1][0]);
  c[0][1] = wmma_bf16(a0, b1, c[0][1]);
  c[1][1] = wmma_bf16(a1, b1, c[1][1]);
  c[0][2] = wmma_bf16(a0, b2, c[0][2]);
  c[1][2] = wmma_bf16(a1, b2, c[1][2]);
  c[0][3] = wmma_bf16(a0, b3, c[0][3]);
  c[1][3] = wmma_bf16(a1, b3, c[1][3]);
}

// butterfly reductions over the 16-lane half-group (masks stay inside the half)
__device__ __forceinline__ float rmax16(float t) {
  t = fmaxf(t, __shfl_xor(t, 1, 32));
  t = fmaxf(t, __shfl_xor(t, 2, 32));
  t = fmaxf(t, __shfl_xor(t, 4, 32));
  t = fmaxf(t, __shfl_xor(t, 8, 32));
  return t;
}
__device__ __forceinline__ float rsum16(float t) {
  t += __shfl_xor(t, 1, 32);
  t += __shfl_xor(t, 2, 32);
  t += __shfl_xor(t, 4, 32);
  t += __shfl_xor(t, 8, 32);
  return t;
}

// ---- conversion kernels ----------------------------------------------------
__global__ __launch_bounds__(256) void cvt_f32_to_bf16(const float* __restrict__ in,
                                                       bf16_t* __restrict__ out, int n) {
  int i = blockIdx.x * blockDim.x + threadIdx.x;
  if (i < n) out[i] = (bf16_t)in[i];
}

// in[R][C] (f32) -> out[C][R] (bf16)
__global__ __launch_bounds__(256) void transpose_cvt(const float* __restrict__ in,
                                                     bf16_t* __restrict__ out, int R, int C) {
  int i = blockIdx.x * blockDim.x + threadIdx.x;
  if (i < R * C) {
    int r = i / C, c = i - r * C;
    out[(size_t)c * R + r] = (bf16_t)in[i];
  }
}

// ---- 32x64 GEMM core, two-stage ping-pong pipeline -------------------------
// Fragment set A covers even k-steps, set B odd k-steps. No register rotation:
// setA's refill loads are in flight during setB's 8 WMMAs and vice versa.
// unroll(disable) pins the schedule: exactly 2 fragment sets live (~180 VGPRs
// total with accumulators) -> no VGPR-MSB bank switching, no accumulator movs.
// Requires K % 64 == 0 and K >= 128.
__device__ __forceinline__ void gemm_32x64(const bf16_t* Arow0, const bf16_t* Arow1,
                                           const bf16_t* Bt0, const bf16_t* Bt1,
                                           const bf16_t* Bt2, const bf16_t* Bt3,
                                           int K, int lane, v8f c[2][4]) {
  v16bf a0A = load_a_frag(Arow0, K, 0, lane);
  v16bf a1A = load_a_frag(Arow1, K, 0, lane);
  v16bf b0A = load_bt_frag(Bt0, K, 0, lane);
  v16bf b1A = load_bt_frag(Bt1, K, 0, lane);
  v16bf b2A = load_bt_frag(Bt2, K, 0, lane);
  v16bf b3A = load_bt_frag(Bt3, K, 0, lane);
  v16bf a0B = load_a_frag(Arow0, K, 32, lane);
  v16bf a1B = load_a_frag(Arow1, K, 32, lane);
  v16bf b0B = load_bt_frag(Bt0, K, 32, lane);
  v16bf b1B = load_bt_frag(Bt1, K, 32, lane);
  v16bf b2B = load_bt_frag(Bt2, K, 32, lane);
  v16bf b3B = load_bt_frag(Bt3, K, 32, lane);

  int k0 = 0;
#pragma clang loop unroll(disable)
  for (; k0 < K - 96; k0 += 64) {
    // consume k0 (set A), refill set A with k0+64
    mma8(c, a0A, a1A, b0A, b1A, b2A, b3A);
    a0A = load_a_frag(Arow0, K, k0 + 64, lane);
    a1A = load_a_frag(Arow1, K, k0 + 64, lane);
    b0A = load_bt_frag(Bt0, K, k0 + 64, lane);
    b1A = load_bt_frag(Bt1, K, k0 + 64, lane);
    b2A = load_bt_frag(Bt2, K, k0 + 64, lane);
    b3A = load_bt_frag(Bt3, K, k0 + 64, lane);
    __builtin_prefetch(Arow0 + k0 + 128, 0, 3);
    __builtin_prefetch(Arow1 + k0 + 128, 0, 3);
    // consume k0+32 (set B), refill set B with k0+96
    mma8(c, a0B, a1B, b0B, b1B, b2B, b3B);
    a0B = load_a_frag(Arow0, K, k0 + 96, lane);
    a1B = load_a_frag(Arow1, K, k0 + 96, lane);
    b0B = load_bt_frag(Bt0, K, k0 + 96, lane);
    b1B = load_bt_frag(Bt1, K, k0 + 96, lane);
    b2B = load_bt_frag(Bt2, K, k0 + 96, lane);
    b3B = load_bt_frag(Bt3, K, k0 + 96, lane);
  }
  // drain: set A holds K-64, set B holds K-32
  mma8(c, a0A, a1A, b0A, b1A, b2A, b3A);
  mma8(c, a0B, a1B, b0B, b1B, b2B, b3B);
}

// ---- QKV GEMM: [8192,1024] x [1024,3072] -> scatter into Q/K/V^T -----------
// wave tile 32x64; grid: 256 M-tiles * 48 N-tiles = 12288 waves = 1536 blocks
__global__ __launch_bounds__(256) void qkv_gemm_kernel(const bf16_t* __restrict__ xb,
                                                       const bf16_t* __restrict__ wqkvT,
                                                       bf16_t* __restrict__ Qb,
                                                       bf16_t* __restrict__ Kb,
                                                       bf16_t* __restrict__ Vt) {
  const int lane = threadIdx.x & 31;
  const int wave = blockIdx.x * (blockDim.x >> 5) + (threadIdx.x >> 5);
  const int tm = wave & 255;
  const int tn = wave >> 8;
  const int m0 = tm * 32, n0 = tn * 64;

  v8f zero = {};
  v8f c[2][4];
#pragma unroll
  for (int i = 0; i < 2; ++i)
#pragma unroll
    for (int j = 0; j < 4; ++j) c[i][j] = zero;

  gemm_32x64(xb + (size_t)m0 * DIM, xb + (size_t)(m0 + 16) * DIM,
             wqkvT + (size_t)(n0 +  0) * DIM, wqkvT + (size_t)(n0 + 16) * DIM,
             wqkvT + (size_t)(n0 + 32) * DIM, wqkvT + (size_t)(n0 + 48) * DIM,
             DIM, lane, c);

  // scatter: this wave's 64 columns live in exactly one (component, head)
  const int comp = n0 >> 10;          // 0=Q, 1=K, 2=V
  const int h    = (n0 & 1023) >> 6;  // head
  const int r16 = lane & 15, hi = lane >> 4;
#pragma unroll
  for (int i = 0; i < 2; ++i)
#pragma unroll
    for (int j = 0; j < 4; ++j)
#pragma unroll
      for (int v = 0; v < 8; ++v) {
        float val = c[i][j][v];
        int d  = 16 * j + r16;                  // 0..63 within head
        int mg = m0 + 16 * i + hi * 8 + v;      // global row
        int bb = mg >> 11, nn = mg & 2047;
        bf16_t bv = (bf16_t)val;
        if (comp == 0)
          Qb[((size_t)(bb * HEADS + h) * SEQ + nn) * HD + d] = bv;
        else if (comp == 1)
          Kb[((size_t)(bb * HEADS + h) * SEQ + nn) * HD + d] = bv;
        else
          Vt[((size_t)(bb * HEADS + h) * HD + d) * SEQ + nn] = bv;
      }
}

// ---- Flash attention: one wave per (b,h, 16-query tile) --------------------
// grid: 4*16*128 = 8192 waves = 1024 blocks of 256 threads
__global__ __launch_bounds__(256) void attn_kernel(const bf16_t* __restrict__ Qb,
                                                   const bf16_t* __restrict__ Kb,
                                                   const bf16_t* __restrict__ Vt,
                                                   bf16_t* __restrict__ Ob) {
  __shared__ bf16_t ldsP[8][16 * 32];   // per-wave P tile, bf16 row-major [16][32]
  const int wid  = threadIdx.x >> 5;
  const int lane = threadIdx.x & 31;
  const int wave = blockIdx.x * 8 + wid;
  const int qt = wave & 127;            // 128 query tiles
  const int bh = wave >> 7;             // b*16 + h
  const int b  = bh >> 4;
  const int h  = bh & 15;
  const int q0 = qt * 16;

  const bf16_t* Qh = Qb + (size_t)bh * SEQ * HD;
  const bf16_t* Kh = Kb + (size_t)bh * SEQ * HD;
  const bf16_t* Vh = Vt + (size_t)bh * HD * SEQ;
  bf16_t* myP = &ldsP[wid][0];

  const float scale = 0.125f;           // 1/sqrt(64)

  // Q A-fragments (hd=64 -> two 16x32 fragments), loaded once
  v16bf aq0 = load_a_frag(Qh + (size_t)q0 * HD, HD, 0,  lane);
  v16bf aq1 = load_a_frag(Qh + (size_t)q0 * HD, HD, 32, lane);

  v8f zero = {};
  v8f o[4];
  float mrow[8], lrow[8];
#pragma unroll
  for (int j = 0; j < 4; ++j) o[j] = zero;
#pragma unroll
  for (int v = 0; v < 8; ++v) { mrow[v] = -1e30f; lrow[v] = 0.0f; }

  const int r16 = lane & 15, hi = lane >> 4;

#pragma clang loop unroll(disable)
  for (int kt = 0; kt < SEQ; kt += 32) {
    if (kt + 32 < SEQ) {
      __builtin_prefetch(Kh + (size_t)(kt + 32) * HD, 0, 3);
      __builtin_prefetch(Vh + kt + 32, 0, 3);
    }
    // S tile 16x32 = Q(16x64) . K^T : two 16x16 C frags, 2 k-steps each
    v16bf bk00 = load_bt_frag(Kh + (size_t)kt * HD, HD, 0,  lane);
    v16bf bk01 = load_bt_frag(Kh + (size_t)kt * HD, HD, 32, lane);
    v16bf bk10 = load_bt_frag(Kh + (size_t)(kt + 16) * HD, HD, 0,  lane);
    v16bf bk11 = load_bt_frag(Kh + (size_t)(kt + 16) * HD, HD, 32, lane);
    v8f s0 = zero, s1 = zero;
    s0 = wmma_bf16(aq0, bk00, s0);
    s1 = wmma_bf16(aq0, bk10, s1);
    s0 = wmma_bf16(aq1, bk01, s0);
    s1 = wmma_bf16(aq1, bk11, s1);

    // V fragments are independent of the softmax: issue the loads now so they
    // are in flight during the VALU/transcendental-heavy softmax below.
    v16bf bv0 = load_bt_frag(Vh + (size_t) 0 * SEQ, SEQ, kt, lane);
    v16bf bv1 = load_bt_frag(Vh + (size_t)16 * SEQ, SEQ, kt, lane);
    v16bf bv2 = load_bt_frag(Vh + (size_t)32 * SEQ, SEQ, kt, lane);
    v16bf bv3 = load_bt_frag(Vh + (size_t)48 * SEQ, SEQ, kt, lane);

    // online softmax over this 32-key slab (row = hi*8 + v)
    float mnew[8], alpha[8];
#pragma unroll
    for (int v = 0; v < 8; ++v) {
      float a = s0[v] * scale;
      float bvv = s1[v] * scale;
      s0[v] = a; s1[v] = bvv;
      float t = rmax16(fmaxf(a, bvv));
      mnew[v]  = fmaxf(mrow[v], t);
      alpha[v] = __expf(mrow[v] - mnew[v]);
      mrow[v]  = mnew[v];
    }
#pragma unroll
    for (int v = 0; v < 8; ++v) {
      float p0 = __expf(s0[v] - mnew[v]);
      float p1 = __expf(s1[v] - mnew[v]);
      s0[v] = p0; s1[v] = p1;
      float rs = rsum16(p0 + p1);
      lrow[v] = lrow[v] * alpha[v] + rs;
    }
#pragma unroll
    for (int j = 0; j < 4; ++j)
#pragma unroll
      for (int v = 0; v < 8; ++v) o[j][v] *= alpha[v];

    // P (C layout) -> LDS bf16 [16][32]
#pragma unroll
    for (int v = 0; v < 8; ++v) {
      int row = hi * 8 + v;
      myP[row * 32 + r16]      = (bf16_t)s0[v];
      myP[row * 32 + 16 + r16] = (bf16_t)s1[v];
    }

    // reload as A fragment and accumulate O += P . V (V^T rows = d)
    v16bf ap = load_a_frag(myP, 32, 0, lane);
    o[0] = wmma_bf16(ap, bv0, o[0]);
    o[1] = wmma_bf16(ap, bv1, o[1]);
    o[2] = wmma_bf16(ap, bv2, o[2]);
    o[3] = wmma_bf16(ap, bv3, o[3]);
  }

  // epilogue: O /= l, store bf16 into [B][N][D] layout for the proj GEMM
#pragma unroll
  for (int j = 0; j < 4; ++j)
#pragma unroll
    for (int v = 0; v < 8; ++v) {
      float val = o[j][v] / lrow[v];
      int row = q0 + hi * 8 + v;
      int col = h * HD + 16 * j + r16;
      Ob[((size_t)b * SEQ + row) * DIM + col] = (bf16_t)val;
    }
}

// ---- Projection GEMM: [8192,1024] x [1024,1024] + bias -> f32 out ----------
// wave tile 32x64; grid: 256 * 16 = 4096 waves = 512 blocks
__global__ __launch_bounds__(256) void proj_gemm_kernel(const bf16_t* __restrict__ Ob,
                                                        const bf16_t* __restrict__ wprojT,
                                                        const float* __restrict__ bias,
                                                        float* __restrict__ out) {
  const int lane = threadIdx.x & 31;
  const int wave = blockIdx.x * (blockDim.x >> 5) + (threadIdx.x >> 5);
  const int tm = wave & 255;
  const int tn = wave >> 8;
  const int m0 = tm * 32, n0 = tn * 64;

  v8f zero = {};
  v8f c[2][4];
#pragma unroll
  for (int i = 0; i < 2; ++i)
#pragma unroll
    for (int j = 0; j < 4; ++j) c[i][j] = zero;

  gemm_32x64(Ob + (size_t)m0 * DIM, Ob + (size_t)(m0 + 16) * DIM,
             wprojT + (size_t)(n0 +  0) * DIM, wprojT + (size_t)(n0 + 16) * DIM,
             wprojT + (size_t)(n0 + 32) * DIM, wprojT + (size_t)(n0 + 48) * DIM,
             DIM, lane, c);

  const int r16 = lane & 15, hi = lane >> 4;
#pragma unroll
  for (int i = 0; i < 2; ++i)
#pragma unroll
    for (int j = 0; j < 4; ++j)
#pragma unroll
      for (int v = 0; v < 8; ++v) {
        int col = n0 + 16 * j + r16;
        int mg  = m0 + 16 * i + hi * 8 + v;
        out[(size_t)mg * DIM + col] = c[i][j][v] + bias[col];
      }
}

// ---- launch ----------------------------------------------------------------
extern "C" void kernel_launch(void* const* d_in, const int* in_sizes, int n_in,
                              void* d_out, int out_size, void* d_ws, size_t ws_size,
                              hipStream_t stream) {
  const float* x      = (const float*)d_in[0];
  const float* W_qkv  = (const float*)d_in[1];
  const float* W_proj = (const float*)d_in[2];
  const float* b_proj = (const float*)d_in[3];
  float* out = (float*)d_out;

  char* ws = (char*)d_ws;
  bf16_t* xb     = (bf16_t*)(ws + 0);           // 16 MB  [8192][1024]
  bf16_t* wqkvT  = (bf16_t*)(ws + 16777216);    //  6 MB  [3072][1024]
  bf16_t* wprojT = (bf16_t*)(ws + 23068672);    //  2 MB  [1024][1024]
  bf16_t* Qb     = (bf16_t*)(ws + 25165824);    // 16 MB  [B][H][N][hd]
  bf16_t* Kb     = (bf16_t*)(ws + 41943040);    // 16 MB  [B][H][N][hd]
  bf16_t* Vt     = (bf16_t*)(ws + 58720256);    // 16 MB  [B][H][hd][N]
  bf16_t* Ob     = (bf16_t*)(ws + 75497472);    // 16 MB  [B][N][D]

  const int nx = MROWS * DIM;                   // 8388608
  cvt_f32_to_bf16<<<(nx + 255) / 256, 256, 0, stream>>>(x, xb, nx);
  transpose_cvt<<<(DIM * N3 + 255) / 256, 256, 0, stream>>>(W_qkv, wqkvT, DIM, N3);
  transpose_cvt<<<(DIM * DIM + 255) / 256, 256, 0, stream>>>(W_proj, wprojT, DIM, DIM);

  qkv_gemm_kernel<<<1536, 256, 0, stream>>>(xb, wqkvT, Qb, Kb, Vt);
  attn_kernel<<<1024, 256, 0, stream>>>(Qb, Kb, Vt, Ob);
  proj_gemm_kernel<<<512, 256, 0, stream>>>(Ob, wprojT, b_proj, out);
}
